// MultiScaleRetention_69879117906737
// MI455X (gfx1250) — compile-verified, matrix operand order checked by reading, not run on hardware
//
#include <hip/hip_runtime.h>
#include <hip/hip_bf16.h>
#include <math.h>

typedef __attribute__((ext_vector_type(16))) _Float16 v16h;
typedef __attribute__((ext_vector_type(8)))  _Float16 v8h;
typedef __attribute__((ext_vector_type(4)))  _Float16 v4h;
typedef __attribute__((ext_vector_type(8)))  float    v8f;
typedef __attribute__((ext_vector_type(4)))  float    v4f;

#define B_  4
#define S_  2048
#define D_  512
#define H_  8
#define DK_ 64

// Wave32 16-bit WMMA A/B fragment (ISA 7.12.2): per lane (row/col = lane&15),
// VGPRs 0-3 hold K = 8*laneHi + 0..7 (contiguous), VGPRs 4-7 hold
// K = 16 + 8*laneHi + 0..7. A row-major f16 source fragment = two 16B loads.
__device__ __forceinline__ v16h frag_from_f16_row(const _Float16* __restrict__ p,
                                                  int laneHi) {
  v8h lo = *(const v8h*)(p + laneHi * 8);        // K =  8*laneHi + 0..7
  v8h hi = *(const v8h*)(p + 16 + laneHi * 8);   // K = 16 + 8*laneHi + 0..7
  v16h f;
#pragma unroll
  for (int j = 0; j < 8; ++j) { f[j] = lo[j]; f[8 + j] = hi[j]; }
  return f;
}

// ---------------------------------------------------------------------------
// Kernel 0: `out` is exactly zero (GroupNorm with one element per group).
// ---------------------------------------------------------------------------
__global__ __launch_bounds__(256) void zero_out_kernel(float* __restrict__ p) {
  size_t i = ((size_t)blockIdx.x * blockDim.x + threadIdx.x) * 4;
  v4f z = {};
  __builtin_nontemporal_store(z, (v4f*)(p + i));
}

// ---------------------------------------------------------------------------
// Prep A: x (f32, read once -> NT load) -> xh (f16, re-read 8x, temporal).
// ---------------------------------------------------------------------------
__global__ __launch_bounds__(256) void cvt_x_kernel(const float* __restrict__ x,
                                                    _Float16* __restrict__ xh) {
  size_t i = ((size_t)blockIdx.x * blockDim.x + threadIdx.x) * 4;
  v4f v = __builtin_nontemporal_load((const v4f*)(x + i));
  v4h hv;
#pragma unroll
  for (int j = 0; j < 4; ++j) hv[j] = (_Float16)v[j];
  *(v4h*)(xh + i) = hv;
}

// ---------------------------------------------------------------------------
// Prep B: W[h][d][n] (f32) -> WT[h][n][d] (f16). Makes the WMMA B-fragment
// contiguous in K(=d): two b128 loads per fragment in the projection GEMM.
// Coalesced reads; 2-byte scattered writes merge in L2 (0.5 MB, resident).
// ---------------------------------------------------------------------------
__global__ __launch_bounds__(256) void transpose_w_kernel(
    const float* __restrict__ Wq, const float* __restrict__ Wk,
    _Float16* __restrict__ WqT, _Float16* __restrict__ WkT) {
  const float*    src = blockIdx.y ? Wk : Wq;
  _Float16*       dst = blockIdx.y ? WkT : WqT;
  unsigned g = blockIdx.x * blockDim.x + threadIdx.x;   // < H*D*DK = 262144
  int n = g & (DK_ - 1);
  int d = (g >> 6) & (D_ - 1);
  int h = g >> 15;
  float v = __builtin_nontemporal_load(src + g);
  dst[((size_t)h * DK_ + n) * D_ + d] = (_Float16)v;
}

// ---------------------------------------------------------------------------
// Kernel 1: Q = x @ W_Q[h], K = x @ W_K[h] -> f16 workspace (L2-resident).
// 128 threads = 4 waves, each owns 16 rows of a 64-row tile. K-loop over
// D=512 in chunks of 32; all fragments are pure b128 loads (pre-packed f16).
// ---------------------------------------------------------------------------
__global__ __launch_bounds__(128) void proj_qk_kernel(
    const _Float16* __restrict__ xh, const _Float16* __restrict__ WqT,
    const _Float16* __restrict__ WkT, _Float16* __restrict__ Qw,
    _Float16* __restrict__ Kw) {
  const int lane   = threadIdx.x & 31;
  const int wave   = threadIdx.x >> 5;
  const int laneHi = (lane >> 4) & 1;
  const int ncol   = lane & 15;
  const int h      = blockIdx.y;
  const int row0   = blockIdx.x * 64 + wave * 16;   // row into flattened [B*S]
  const int myrow  = row0 + (lane & 15);

  const _Float16* xrow = xh + (size_t)myrow * D_;

  v8f accQ[4] = {};
  v8f accK[4] = {};

  for (int d0 = 0; d0 < D_; d0 += 32) {
    v16h a = frag_from_f16_row(xrow + d0, laneHi);
#pragma unroll
    for (int nt = 0; nt < 4; ++nt) {
      const size_t col = (size_t)h * DK_ + nt * 16 + ncol;   // output column
      v16h bq = frag_from_f16_row(WqT + col * D_ + d0, laneHi);
      accQ[nt] = __builtin_amdgcn_wmma_f32_16x16x32_f16(
          false, a, false, bq, (short)0, accQ[nt], false, false);
      v16h bk = frag_from_f16_row(WkT + col * D_ + d0, laneHi);
      accK[nt] = __builtin_amdgcn_wmma_f32_16x16x32_f16(
          false, a, false, bk, (short)0, accK[nt], false, false);
    }
  }

  // C/D layout: lane -> N = lane&15, M = r + 8*laneHi. Store f32->f16.
#pragma unroll
  for (int nt = 0; nt < 4; ++nt) {
#pragma unroll
    for (int r = 0; r < 8; ++r) {
      int m  = r + laneHi * 8;
      int rg = row0 + m;
      int b  = rg >> 11;          // / S_
      int s  = rg & (S_ - 1);
      size_t qi = (((size_t)b * H_ + h) * S_ + s) * DK_ + nt * 16 + ncol;
      Qw[qi] = (_Float16)accQ[nt][r];
      Kw[qi] = (_Float16)accK[nt][r];
    }
  }
}

// ---------------------------------------------------------------------------
// Kernel 2: A[b,h,s,t] = (Q[s,:].K[t,:]) * gamma_h^(s-t) for s>=t else 0.
// 64x64 tiles; above-diagonal tiles NT zero-filled; others run 2 WMMAs per
// 16x16 sub-tile (K=64). All A traffic is non-temporal: 537 MB streamed once,
// must not evict the L2-resident Q/K (16.8 MB vs 192 MB L2).
// ---------------------------------------------------------------------------
__global__ __launch_bounds__(128) void retention_scores_kernel(
    const _Float16* __restrict__ Qw, const _Float16* __restrict__ Kw,
    float* __restrict__ A) {
  const int lane   = threadIdx.x & 31;
  const int wave   = threadIdx.x >> 5;
  const int laneHi = (lane >> 4) & 1;
  const int bh     = blockIdx.z;
  const int h      = bh & (H_ - 1);
  const int t0     = blockIdx.x * 64;
  const int s0     = blockIdx.y * 64;

  float* __restrict__ Abh = A + (size_t)bh * S_ * S_;

  if (t0 > s0 + 63) {  // entire tile above the causal diagonal -> zeros
    v4f z = {};
#pragma unroll
    for (int i = 0; i < 8; ++i) {
      int fl  = threadIdx.x + i * 128;   // 0..1023 float4 slots in tile
      int row = fl >> 4;
      int c4  = fl & 15;
      __builtin_nontemporal_store(
          z, (v4f*)(Abh + (size_t)(s0 + row) * S_ + t0 + c4 * 4));
    }
    return;
  }

  // Per-head decay: gamma_h = 1 - exp(linspace(log(1/32), log(1/512), 8))[h]
  const float l0    = logf(1.0f / 32.0f);
  const float dl    = (logf(1.0f / 512.0f) - l0) * (1.0f / 7.0f);
  const float gam   = 1.0f - expf(l0 + (float)h * dl);
  const float log2g = log2f(gam);
  float garr[8];                        // gamma^r, r = 0..7
  garr[0] = 1.0f;
#pragma unroll
  for (int r = 1; r < 8; ++r) garr[r] = garr[r - 1] * gam;

  const int srow = s0 + wave * 16 + (lane & 15);
  const _Float16* qrow = Qw + ((size_t)bh * S_ + srow) * DK_;
  v16h a0 = frag_from_f16_row(qrow, laneHi);        // K = 0..31
  v16h a1 = frag_from_f16_row(qrow + 32, laneHi);   // K = 32..63

  const int sbase = s0 + wave * 16 + 8 * laneHi;    // M-row base for this lane

#pragma unroll
  for (int nt = 0; nt < 4; ++nt) {
    const int tcol = t0 + nt * 16 + (lane & 15);
    const _Float16* krow = Kw + ((size_t)bh * S_ + tcol) * DK_;
    v16h b0 = frag_from_f16_row(krow, laneHi);
    v16h b1 = frag_from_f16_row(krow + 32, laneHi);

    v8f acc = {};
    acc = __builtin_amdgcn_wmma_f32_16x16x32_f16(
        false, a0, false, b0, (short)0, acc, false, false);
    acc = __builtin_amdgcn_wmma_f32_16x16x32_f16(
        false, a1, false, b1, (short)0, acc, false, false);

    const int   t    = tcol;                 // N = lane&15 already included
    const int   d0i  = sbase - t;            // in [-63, 2047]; gamma^d0i finite
    const float base = exp2f((float)d0i * log2g);
#pragma unroll
    for (int r = 0; r < 8; ++r) {
      int   s   = sbase + r;
      float dec = ((d0i + r) >= 0) ? base * garr[r] : 0.0f;
      __builtin_nontemporal_store(acc[r] * dec, Abh + (size_t)s * S_ + t);
    }
  }
}

// ---------------------------------------------------------------------------
extern "C" void kernel_launch(void* const* d_in, const int* in_sizes, int n_in,
                              void* d_out, int out_size, void* d_ws, size_t ws_size,
                              hipStream_t stream) {
  const float* x  = (const float*)d_in[0];
  const float* Wq = (const float*)d_in[1];
  const float* Wk = (const float*)d_in[2];
  // d_in[3..5] (W_V, W_G, W_O) are mathematically irrelevant: GroupNorm with
  // one element per group forces out == 0 exactly.

  float* out = (float*)d_out;
  float* A   = out + (size_t)B_ * S_ * D_;   // A follows out in the flat tuple

  // Workspace layout (f16): Qw | Kw | xh | WqT | WkT  (~26 MB total)
  const size_t NQK = (size_t)B_ * H_ * S_ * DK_;   // 4,194,304
  const size_t NX  = (size_t)B_ * S_ * D_;         // 4,194,304
  const size_t NW  = (size_t)H_ * DK_ * D_;        // 262,144
  _Float16* Qw  = (_Float16*)d_ws;
  _Float16* Kw  = Qw + NQK;
  _Float16* xh  = Kw + NQK;
  _Float16* WqT = xh + NX;
  _Float16* WkT = WqT + NW;

  // 0) out = 0 (exact)
  zero_out_kernel<<<(unsigned)(NX / 4 / 256), 256, 0, stream>>>(out);

  // 1) one-shot repacks: x -> f16, W -> f16 transposed [h][n][d]
  cvt_x_kernel<<<(unsigned)(NX / 4 / 256), 256, 0, stream>>>(x, xh);
  dim3 gt((unsigned)(NW / 256), 2);
  transpose_w_kernel<<<gt, 256, 0, stream>>>(Wq, Wk, WqT, WkT);

  // 2) Q/K projections (all-b128 fragment loads)
  dim3 g1(B_ * S_ / 64, H_);
  proj_qk_kernel<<<g1, 128, 0, stream>>>(xh, WqT, WkT, Qw, Kw);

  // 3) decayed retention scores -> A (HBM-write bound: ~537 MB @ 23.3 TB/s)
  dim3 g2(S_ / 64, S_ / 64, B_ * H_);
  retention_scores_kernel<<<g2, 128, 0, stream>>>(Qw, Kw, A);
}
